// CartesianPlaneNonSirenEmbeddingNetwork_19490561589627
// MI455X (gfx1250) — compile-verified, matrix operand order checked by reading, not run on hardware
//
#include <hip/hip_runtime.h>

typedef __attribute__((ext_vector_type(16))) __bf16 v16bf;
typedef __attribute__((ext_vector_type(8)))  __bf16 v8bf;
typedef __attribute__((ext_vector_type(8)))  float  v8f;

#define NW   8            // waves per block
#define BLK  (NW * 32)
#define LDK  136          // padded LDS row stride in bf16 elements (272B, 16B-aligned)
#define CCH  32
#define HH   256
#define WWD  256
#define HW   (HH * WWD)

__launch_bounds__(BLK)
__global__ void triplane_mlp_kernel(
    const float* __restrict__ coords,
    const float* __restrict__ p0,
    const float* __restrict__ p1,
    const float* __restrict__ p2,
    const float* __restrict__ W1g, const float* __restrict__ b1g,
    const float* __restrict__ W2g, const float* __restrict__ b2g,
    const float* __restrict__ W3g, const float* __restrict__ b3g,
    float* __restrict__ out, int totalPts)
{
    __shared__ __align__(16) __bf16 sW1[128 * LDK];
    __shared__ __align__(16) __bf16 sW2[128 * LDK];
    __shared__ float sB1[128];
    __shared__ float sB2[128];
    __shared__ float sW3[128];
    __shared__ __align__(16) __bf16 sF[NW][16 * LDK];

    const int tid = threadIdx.x;

    // ---- cooperative weight preload (bf16, K padded to 128 with zeros) ----
    for (int i = tid; i < 128 * 128; i += BLK) {
        int n = i >> 7, d = i & 127;
        float v1 = (d < 123) ? W1g[n * 123 + d] : 0.0f;
        sW1[n * LDK + d] = (__bf16)v1;
        sW2[n * LDK + d] = (__bf16)W2g[n * 128 + d];
    }
    if (tid < 128) { sB1[tid] = b1g[tid]; sB2[tid] = b2g[tid]; sW3[tid] = W3g[tid]; }
    __syncthreads();

    const int lane = tid & 31;
    const int wv   = tid >> 5;
    const int g    = lane >> 4;   // half-wave id (K-striping group)
    const int m    = lane & 15;   // row (point) / column (neuron) within tile
    const float b3 = b3g[0];

    __bf16* F = &sF[wv][0];

    const int T      = totalPts >> 4;               // 16-point tiles
    const int gw     = blockIdx.x * NW + wv;
    const int stride = gridDim.x * NW;

    for (int tile = gw; tile < T; tile += stride) {
        const int base = tile << 4;
        const int pidx = base + m;

        // ================= sampling + posenc -> bf16 feature row =================
        const float c0 = coords[pidx * 3 + 0];
        const float c1 = coords[pidx * 3 + 1];
        const float c2 = coords[pidx * 3 + 2];
        const int cbase = g << 4;                   // this lane's 16-channel half

        const float uvx[3] = { c0, c1, c0 };
        const float uvy[3] = { c1, c2, c2 };
        const float* const planes[3] = { p0, p1, p2 };

        #pragma unroll
        for (int p = 0; p < 3; ++p) {
            float x = (uvx[p] + 1.0f) * 0.5f * (float)(WWD - 1);
            float y = (uvy[p] + 1.0f) * 0.5f * (float)(HH - 1);
            float fx = floorf(x), fy = floorf(y);
            int ix0 = (int)fx, iy0 = (int)fy;
            int ix1 = ix0 + 1, iy1 = iy0 + 1;
            float wx1 = x - fx, wy1 = y - fy;
            float wx0 = 1.0f - wx1, wy0 = 1.0f - wy1;
            float vx0 = (ix0 >= 0 && ix0 < WWD) ? 1.0f : 0.0f;
            float vx1 = (ix1 >= 0 && ix1 < WWD) ? 1.0f : 0.0f;
            float vy0 = (iy0 >= 0 && iy0 < HH)  ? 1.0f : 0.0f;
            float vy1 = (iy1 >= 0 && iy1 < HH)  ? 1.0f : 0.0f;
            int xc0 = min(max(ix0, 0), WWD - 1), xc1 = min(max(ix1, 0), WWD - 1);
            int yc0 = min(max(iy0, 0), HH - 1),  yc1 = min(max(iy1, 0), HH - 1);
            float w00 = wx0 * wy0 * vx0 * vy0;
            float w10 = wx1 * wy0 * vx1 * vy0;
            float w01 = wx0 * wy1 * vx0 * vy1;
            float w11 = wx1 * wy1 * vx1 * vy1;
            int o00 = yc0 * WWD + xc0;
            int o10 = yc0 * WWD + xc1;
            int o01 = yc1 * WWD + xc0;
            int o11 = yc1 * WWD + xc1;
            const float* pb = planes[p] + cbase * HW;

            // full unroll: 64 gathers in flight, constant channel stride folds
            // into the global_load 24-bit immediate offset (c*HW*4 < 2^23)
            __bf16 t[16];
            #pragma unroll
            for (int c = 0; c < 16; ++c) {
                const float* q = pb + c * HW;
                float v = w00 * q[o00] + w10 * q[o10] + w01 * q[o01] + w11 * q[o11];
                t[c] = (__bf16)v;
            }
            v8bf* dst = (v8bf*)(F + m * LDK + p * 32 + cbase);   // 16B-aligned
            dst[0] = *(v8bf*)&t[0];
            dst[1] = *(v8bf*)&t[8];
        }
        // posenc: [x(3), sin f0(3), cos f0(3), ... sin f3(3), cos f3(3)] at d=96..122
        // g==0 lane writes d=96..111, g==1 lane writes d=112..127 (incl. zero pad)
        {
            float pe[27];
            pe[0] = c0; pe[1] = c1; pe[2] = c2;
            float f = 1.0f;
            #pragma unroll
            for (int k = 0; k < 4; ++k) {
                pe[3 + 6 * k + 0] = __sinf(c0 * f);
                pe[3 + 6 * k + 1] = __sinf(c1 * f);
                pe[3 + 6 * k + 2] = __sinf(c2 * f);
                pe[3 + 6 * k + 3] = __cosf(c0 * f);
                pe[3 + 6 * k + 4] = __cosf(c1 * f);
                pe[3 + 6 * k + 5] = __cosf(c2 * f);
                f *= 2.0f;
            }
            __bf16 t[16];
            if (g == 0) {
                #pragma unroll
                for (int j = 0; j < 16; ++j) t[j] = (__bf16)pe[j];
            } else {
                #pragma unroll
                for (int j = 0; j < 11; ++j) t[j] = (__bf16)pe[16 + j];
                #pragma unroll
                for (int j = 11; j < 16; ++j) t[j] = (__bf16)0.0f;
            }
            v8bf* dst = (v8bf*)(F + m * LDK + 96 + g * 16);       // 16B-aligned
            dst[0] = *(v8bf*)&t[0];
            dst[1] = *(v8bf*)&t[8];
        }
        // same-wave LDS ops are in-order: no barrier needed before fragment loads

        // ================= layer 1: h1 = relu(X @ W1^T + b1) =================
        v8f acc[8];
        #pragma unroll
        for (int j = 0; j < 8; ++j)
            acc[j] = (v8f){0.f,0.f,0.f,0.f,0.f,0.f,0.f,0.f};

        #pragma unroll
        for (int ks = 0; ks < 4; ++ks) {
            const __bf16* ap = F + m * LDK + ks * 32 + g * 8;
            v16bf a;
            *((v8bf*)&a)     = *(const v8bf*)(ap);
            *((v8bf*)&a + 1) = *(const v8bf*)(ap + 16);
            #pragma unroll
            for (int j = 0; j < 8; ++j) {
                const __bf16* bp = sW1 + (j * 16 + m) * LDK + ks * 32 + g * 8;
                v16bf b;
                *((v8bf*)&b)     = *(const v8bf*)(bp);
                *((v8bf*)&b + 1) = *(const v8bf*)(bp + 16);
                acc[j] = __builtin_amdgcn_wmma_f32_16x16x32_bf16(
                    false, a, false, b, (short)0, acc[j], false, false);
            }
        }
        // epilogue: bias + relu, write h1 back to LDS tile as next A-matrix
        #pragma unroll
        for (int j = 0; j < 8; ++j) {
            float bias = sB1[j * 16 + m];
            #pragma unroll
            for (int r = 0; r < 8; ++r) {
                float h = acc[j][r] + bias;
                h = h > 0.0f ? h : 0.0f;
                F[(r + 8 * g) * LDK + j * 16 + m] = (__bf16)h;
            }
        }

        // ================= layer 2 + fused layer 3 =================
        #pragma unroll
        for (int j = 0; j < 8; ++j)
            acc[j] = (v8f){0.f,0.f,0.f,0.f,0.f,0.f,0.f,0.f};

        #pragma unroll
        for (int ks = 0; ks < 4; ++ks) {
            const __bf16* ap = F + m * LDK + ks * 32 + g * 8;
            v16bf a;
            *((v8bf*)&a)     = *(const v8bf*)(ap);
            *((v8bf*)&a + 1) = *(const v8bf*)(ap + 16);
            #pragma unroll
            for (int j = 0; j < 8; ++j) {
                const __bf16* bp = sW2 + (j * 16 + m) * LDK + ks * 32 + g * 8;
                v16bf b;
                *((v8bf*)&b)     = *(const v8bf*)(bp);
                *((v8bf*)&b + 1) = *(const v8bf*)(bp + 16);
                acc[j] = __builtin_amdgcn_wmma_f32_16x16x32_bf16(
                    false, a, false, b, (short)0, acc[j], false, false);
            }
        }
        // bias + relu + dot with W3 (this lane owns column N = j*16 + m)
        float op[8];
        #pragma unroll
        for (int r = 0; r < 8; ++r) op[r] = 0.0f;
        #pragma unroll
        for (int j = 0; j < 8; ++j) {
            float bias = sB2[j * 16 + m];
            float w3   = sW3[j * 16 + m];
            #pragma unroll
            for (int r = 0; r < 8; ++r) {
                float h = acc[j][r] + bias;
                h = h > 0.0f ? h : 0.0f;
                op[r] += h * w3;
            }
        }
        // butterfly-reduce across the 16 lanes of each half (masks stay in-half)
        #pragma unroll
        for (int s = 1; s < 16; s <<= 1) {
            #pragma unroll
            for (int r = 0; r < 8; ++r)
                op[r] += __shfl_xor(op[r], s, 32);
        }
        if (m == 0) {
            #pragma unroll
            for (int r = 0; r < 8; ++r)
                out[base + 8 * g + r] = op[r] + b3;
        }
    }
}

extern "C" void kernel_launch(void* const* d_in, const int* in_sizes, int n_in,
                              void* d_out, int out_size, void* d_ws, size_t ws_size,
                              hipStream_t stream) {
    const float* coords = (const float*)d_in[0];
    const float* p0     = (const float*)d_in[1];
    const float* p1     = (const float*)d_in[2];
    const float* p2     = (const float*)d_in[3];
    const float* W1     = (const float*)d_in[4];
    const float* b1     = (const float*)d_in[5];
    const float* W2     = (const float*)d_in[6];
    const float* b2     = (const float*)d_in[7];
    const float* W3     = (const float*)d_in[8];
    const float* b3     = (const float*)d_in[9];
    float* out = (float*)d_out;

    int totalPts = in_sizes[0] / 3;          // B*N = 1,048,576
    int grid = 4096;                         // grid-stride: 2 tiles per wave
    triplane_mlp_kernel<<<grid, BLK, 0, stream>>>(
        coords, p0, p1, p2, W1, b1, W2, b2, W3, b3, out, totalPts);
}
